// MotionCorrect_non_rigid_2d_10093173145864
// MI455X (gfx1250) — matrix operand order repaired and checked by reading, not compiled, not run on hardware
//
#include <hip/hip_runtime.h>
#include <math.h>

typedef __attribute__((ext_vector_type(16))) _Float16 v16h;
typedef __attribute__((ext_vector_type(8)))  _Float16 v8h;
typedef __attribute__((ext_vector_type(8)))  float    v8f;

#define WMMA_F32_F16(A, B, C) \
  __builtin_amdgcn_wmma_f32_16x16x32_f16(false, (A), false, (B), (short)0, (C), false, false)

__device__ __forceinline__ int iclampi(int x, int lo, int hi) {
  return x < lo ? lo : (x > hi ? hi : x);
}

// ---------------------------------------------------------------------------
// f16 DFT bases: C = cos(2pi jk/N), S = sin, Sn = -sin  (all symmetric NxN)
// ---------------------------------------------------------------------------
__global__ void k_build_dft16(_Float16* __restrict__ C, _Float16* __restrict__ S,
                              _Float16* __restrict__ Sn, int N, int total) {
  int idx = blockIdx.x * blockDim.x + threadIdx.x;
  if (idx >= total) return;
  int j = idx / N, k = idx % N;
  int r = (int)(((long long)j * (long long)k) % (long long)N);
  float th = 6.28318530717958647692f * ((float)r / (float)N);
  float c = cosf(th), s = sinf(th);
  C[idx] = (_Float16)c;
  S[idx] = (_Float16)s;
  Sn[idx] = (_Float16)(-s);
}

__global__ void k_f32_to_f16(const float* __restrict__ src, _Float16* __restrict__ dst,
                             size_t total) {
  size_t idx = (size_t)blockIdx.x * blockDim.x + threadIdx.x;
  if (idx >= total) return;
  dst[idx] = (_Float16)src[idx];
}

// ---------------------------------------------------------------------------
// DFT-pass GEMM via WMMA, f16 operands, f32 accumulate.
//   effA = Ar + i*Ai  (batched, M x K row-major, K contiguous; CPLXA)
//   effB = C + i*bsign*S (symmetric DFT basis, read transposed = itself)
//     accR = Ar*Bc + Ai*Bs1   (Bs1 = -bsign*S, chosen by pointer on host)
//     accI = Ar*Bs2 + Ai*Bc   (Bs2 = +bsign*S; only if CPLXD)
//   D stored TRANSPOSED: D_t[n][m] = scale * (A*B)[m][n]  (contiguous stores)
// One wave computes a 32x32 output tile (2x2 WMMA tiles). All configuration
// is compile-time: the k-loop is branch-free (loads + wmma only).
// ---------------------------------------------------------------------------
template <bool F16OUT, bool CPLXA, bool CPLXD>
__global__ __launch_bounds__(32)
void k_cgemm(const _Float16* __restrict__ Ar, const _Float16* __restrict__ Ai,
             size_t asb, int lda,
             const _Float16* __restrict__ Bc, const _Float16* __restrict__ Bs1,
             const _Float16* __restrict__ Bs2, int ldb,
             void* __restrict__ Drv, void* __restrict__ Div,
             size_t dsb, int ldd, int K, float scale) {
  const int lane = threadIdx.x & 31;
  const int b = blockIdx.z;
  const int m16 = lane & 15;
  const int akb = (lane < 16) ? 0 : 16;   // 16-bit A K-half base
  const int bkb = (lane < 16) ? 0 : 16;   // 16-bit B K-half base

  const _Float16* ar0 = Ar + (size_t)b * asb;
  const _Float16* ai0 = CPLXA ? (Ai + (size_t)b * asb) : ar0;
  const int row0 = blockIdx.y * 32 + m16;
  const _Float16* arp[2] = {ar0 + (size_t)row0 * lda, ar0 + (size_t)(row0 + 16) * lda};
  const _Float16* aip[2] = {ai0 + (size_t)row0 * lda, ai0 + (size_t)(row0 + 16) * lda};
  const int ncol = blockIdx.x * 32 + m16;
  const _Float16* bcp[2] = {Bc + (size_t)ncol * ldb, Bc + (size_t)(ncol + 16) * ldb};
  const _Float16* b1p[2] = {Bs1 + (size_t)ncol * ldb, Bs1 + (size_t)(ncol + 16) * ldb};
  const _Float16* b2p[2] = {Bs2 + (size_t)ncol * ldb, Bs2 + (size_t)(ncol + 16) * ldb};

  v8f aR[2][2] = {};
  v8f aI[2][2] = {};

  for (int kk = 0; kk < K; kk += 32) {
    __builtin_prefetch(arp[0] + kk + 160, 0, 1);   // speculative, safe if OOB
    __builtin_prefetch(arp[1] + kk + 160, 0, 1);
    v16h fAr[2], fAi[2] = {};
#pragma unroll
    for (int mt = 0; mt < 2; ++mt) {
      // A f16 16x32, k order chosen here (B matches): lanes 0-15 hold K {0..15}, lanes 16-31 {16..31}
      v8h c0 = *(const v8h*)(arp[mt] + kk + akb);
      v8h c1 = *(const v8h*)(arp[mt] + kk + akb + 8);
      fAr[mt] = __builtin_shufflevector(c0, c1, 0, 1, 2, 3, 4, 5, 6, 7,
                                        8, 9, 10, 11, 12, 13, 14, 15);
      if (CPLXA) {
        v8h d0 = *(const v8h*)(aip[mt] + kk + akb);
        v8h d1 = *(const v8h*)(aip[mt] + kk + akb + 8);
        fAi[mt] = __builtin_shufflevector(d0, d1, 0, 1, 2, 3, 4, 5, 6, 7,
                                          8, 9, 10, 11, 12, 13, 14, 15);
      }
    }
#pragma unroll
    for (int nt = 0; nt < 2; ++nt) {
      // B f16 32x16, same k order as A: lane holds col n, K contiguous per half-wave.
      // Symmetric basis -> read row n instead of column n: one v16h load.
      v16h fC = *(const v16h*)(bcp[nt] + kk + bkb);
      v16h fS1 = *(const v16h*)(b1p[nt] + kk + bkb);
#pragma unroll
      for (int mt = 0; mt < 2; ++mt) {
        aR[mt][nt] = WMMA_F32_F16(fAr[mt], fC, aR[mt][nt]);
        if (CPLXA) aR[mt][nt] = WMMA_F32_F16(fAi[mt], fS1, aR[mt][nt]);
      }
      if (CPLXD) {
        v16h fS2 = *(const v16h*)(b2p[nt] + kk + bkb);
#pragma unroll
        for (int mt = 0; mt < 2; ++mt) {
          aI[mt][nt] = WMMA_F32_F16(fAr[mt], fS2, aI[mt][nt]);
          if (CPLXA) aI[mt][nt] = WMMA_F32_F16(fAi[mt], fC, aI[mt][nt]);
        }
      }
    }
  }

  // Transposed store: C/D layout gives lane (col n, rows r..r+7) -> contiguous.
  const int mbase = blockIdx.y * 32 + ((lane < 16) ? 0 : 8);
#pragma unroll
  for (int nt = 0; nt < 2; ++nt) {
    const int n = ncol + nt * 16;
#pragma unroll
    for (int mt = 0; mt < 2; ++mt) {
      const int m = mbase + mt * 16;
      const size_t o = (size_t)n * ldd + m;
      v8f vr = aR[mt][nt] * scale;
      if (F16OUT) {
        _Float16* dr = (_Float16*)Drv + (size_t)b * dsb;
        v8h hr;
#pragma unroll
        for (int r = 0; r < 8; ++r) hr[r] = (_Float16)vr[r];
        *(v8h*)(dr + o) = hr;
        if (CPLXD) {
          _Float16* di = (_Float16*)Div + (size_t)b * dsb;
          v8f vi = aI[mt][nt] * scale;
          v8h hi;
#pragma unroll
          for (int r = 0; r < 8; ++r) hi[r] = (_Float16)vi[r];
          *(v8h*)(di + o) = hi;
        }
      } else {
        float* dr = (float*)Drv + (size_t)b * dsb;
        *(v8f*)(dr + o) = vr;
        if (CPLXD) {
          float* di = (float*)Div + (size_t)b * dsb;
          *(v8f*)(di + o) = aI[mt][nt] * scale;
        }
      }
    }
  }
}

// ---------------------------------------------------------------------------
// P = scale * A * conj(Bt) elementwise; f32 in, f16 out (feeds next GEMM)
// ---------------------------------------------------------------------------
__global__ void k_crosspow16(const float* __restrict__ ar, const float* __restrict__ ai,
                             const float* __restrict__ br, const float* __restrict__ bi,
                             _Float16* __restrict__ pr, _Float16* __restrict__ pi,
                             size_t nper, int bmod, float scale, size_t total) {
  size_t idx = (size_t)blockIdx.x * blockDim.x + threadIdx.x;
  if (idx >= total) return;
  size_t b = idx / nper, r = idx % nper;
  size_t tb = (size_t)((int)(b % (size_t)bmod));
  float x = ar[idx], y = ai[idx];
  float u = br[tb * nper + r], v = bi[tb * nper + r];
  pr[idx] = (_Float16)((x * u + y * v) * scale);
  pi[idx] = (_Float16)((y * u - x * v) * scale);
}

// ---------------------------------------------------------------------------
// G = scale * F * exp(-2pi*i*(fH*shx + fW*shy)); f32 in, f16 out
// ---------------------------------------------------------------------------
__global__ void k_phase_mult16(const float* __restrict__ fr, const float* __restrict__ fi,
                               const float* __restrict__ rs,
                               _Float16* __restrict__ gr, _Float16* __restrict__ gi,
                               int H, int W, float scale, size_t total) {
  size_t idx = (size_t)blockIdx.x * blockDim.x + threadIdx.x;
  if (idx >= total) return;
  int j = (int)(idx % W);
  size_t t = idx / W;
  int i = (int)(t % H);
  int b = (int)(t / H);
  float shx = rs[2 * b], shy = rs[2 * b + 1];
  float fh = (float)((i < H / 2) ? i : i - H) / (float)H;
  float fw = (float)((j < W / 2) ? j : j - W) / (float)W;
  float ang = -6.28318530717958647692f * (fh * shx + fw * shy);
  float c = cosf(ang), s = sinf(ang);
  float x = fr[idx], y = fi[idx];
  gr[idx] = (_Float16)((x * c - y * s) * scale);
  gi[idx] = (_Float16)((x * s + y * c) * scale);
}

// ---------------------------------------------------------------------------
// Subpixel peak (11x11 window, log-parabola), matches reference _subpixel
// ---------------------------------------------------------------------------
__device__ __forceinline__ float lg11(const float* ncc, int a, int b) {
  a = iclampi(a, 0, 10);
  b = iclampi(b, 0, 10);
  return logf(ncc[a * 11 + b] + 1e-8f);
}

__device__ void subpixel11(const float* ncc, float lb0, float lb1,
                           float* oshx, float* oshy) {
  int am = 0;
  float best = ncc[0];
  for (int t = 1; t < 121; ++t) {
    if (ncc[t] > best) { best = ncc[t]; am = t; }
  }
  int sx = am / 11, sy = am % 11;
  float lxm = lg11(ncc, sx - 1, sy), lxp = lg11(ncc, sx + 1, sy);
  float lym = lg11(ncc, sx, sy - 1), lyp = lg11(ncc, sx, sy + 1);
  float l0 = 4.0f * lg11(ncc, sx, sy);
  float shx = -((float)sx + lb0) - (lxm - lxp) / (2.0f * lxm - l0 + 2.0f * lxp);
  float shy = -((float)sy + lb1) - (lym - lyp) / (2.0f * lym - l0 + 2.0f * lyp);
  *oshx = shx;
  *oshy = shy;
}

__global__ void k_rigid_subpixel(const float* __restrict__ cr,
                                 const float* __restrict__ ci,
                                 float* __restrict__ rsW, float* __restrict__ rsO) {
  __shared__ float ncc[121];
  const int H = 768, W = 768;
  int b = blockIdx.x;
  const float* crb = cr + (size_t)b * H * W;
  const float* cib = ci + (size_t)b * H * W;
  for (int t = threadIdx.x; t < 121; t += blockDim.x) {
    int di = t / 11 - 5, dj = t % 11 - 5;
    int r = (di + H) % H, c = (dj + W) % W;
    float x = crb[(size_t)r * W + c], y = cib[(size_t)r * W + c];
    ncc[t] = sqrtf(x * x + y * y);
  }
  __syncthreads();
  if (threadIdx.x == 0) {
    float shx, shy;
    subpixel11(ncc, -5.0f, -5.0f, &shx, &shy);
    rsW[2 * b] = shx; rsW[2 * b + 1] = shy;
    rsO[2 * b] = shx; rsO[2 * b + 1] = shy;
  }
}

// ---------------------------------------------------------------------------
// Patch extraction (f16 out): 8x8 grid of 128x128, starts {0,96,...,576,640}
// ---------------------------------------------------------------------------
__global__ void k_extract_patches16(const float* __restrict__ src,
                                    _Float16* __restrict__ pat, int b0,
                                    size_t total) {
  size_t idx = (size_t)blockIdx.x * blockDim.x + threadIdx.x;
  if (idx >= total) return;
  const int P = 128, W = 768;
  int v = (int)(idx % P);
  size_t t1 = idx / P;
  int u = (int)(t1 % P);
  size_t t2 = t1 / P;
  int t = (int)(t2 % 64);
  int bl = (int)(t2 / 64);
  int ix = t >> 3, iy = t & 7;
  int xs = (ix < 7) ? 96 * ix : 640;
  int ys = (iy < 7) ? 96 * iy : 640;
  pat[idx] = (_Float16)src[((size_t)(b0 + bl) * 768 + (size_t)(xs + u)) * W + (ys + v)];
}

__global__ void k_patch_subpixel(const float* __restrict__ cr,
                                 const float* __restrict__ ci,
                                 const float* __restrict__ rs,
                                 float* __restrict__ sxW, float* __restrict__ syW,
                                 float* __restrict__ sxO, float* __restrict__ syO,
                                 int b0) {
  __shared__ float ncc[121];
  const int P = 128;
  int lp = blockIdx.x;
  int bt = b0 * 64 + lp;
  int b = bt / 64;
  float rsx = -rs[2 * b], rsy = -rs[2 * b + 1];
  int lb0 = (int)floorf(rsx - 5.0f);
  int lb1 = (int)floorf(rsy - 5.0f);
  int stx = iclampi(64 + lb0, 0, P - 11);
  int sty = iclampi(64 + lb1, 0, P - 11);
  const float* crb = cr + (size_t)lp * P * P;
  const float* cib = ci + (size_t)lp * P * P;
  for (int t = threadIdx.x; t < 121; t += blockDim.x) {
    int u = t / 11, v = t % 11;
    int r = (stx + u + 64) & 127;
    int c = (sty + v + 64) & 127;
    float x = crb[r * P + c], y = cib[r * P + c];
    ncc[t] = sqrtf(x * x + y * y);
  }
  __syncthreads();
  if (threadIdx.x == 0) {
    float shx, shy;
    subpixel11(ncc, (float)lb0, (float)lb1, &shx, &shy);
    sxW[bt] = -shx; syW[bt] = -shy;
    sxO[bt] = -shx; syO[bt] = -shy;
  }
}

// ---------------------------------------------------------------------------
// Bilinear upsample of 8x8 shift fields + bilinear warp (zero outside)
// ---------------------------------------------------------------------------
__device__ __forceinline__ float sampz(const float* im, int x, int y) {
  if (x < 0 || x >= 768 || y < 0 || y >= 768) return 0.0f;
  return im[(size_t)y * 768 + x];
}

__global__ void k_resize_warp(const float* __restrict__ data,
                              const float* __restrict__ sx8,
                              const float* __restrict__ sy8,
                              float* __restrict__ out, size_t total) {
  size_t idx = (size_t)blockIdx.x * blockDim.x + threadIdx.x;
  if (idx >= total) return;
  const int H = 768, W = 768;
  int j = (int)(idx % W);
  size_t t = idx / W;
  int i = (int)(t % H);
  int b = (int)(t / H);
  const float sc = 8.0f / 768.0f;
  float ui = ((float)i + 0.5f) * sc - 0.5f;
  float uj = ((float)j + 0.5f) * sc - 0.5f;
  float fi0 = floorf(ui), fj0 = floorf(uj);
  float du = ui - fi0, dv = uj - fj0;
  int i0 = iclampi((int)fi0, 0, 7), i1 = iclampi((int)fi0 + 1, 0, 7);
  int j0 = iclampi((int)fj0, 0, 7), j1 = iclampi((int)fj0 + 1, 0, 7);
  const float* gx = sx8 + (size_t)b * 64;
  const float* gy = sy8 + (size_t)b * 64;
  float w00 = (1.0f - du) * (1.0f - dv), w01 = (1.0f - du) * dv;
  float w10 = du * (1.0f - dv), w11 = du * dv;
  float sxv = gx[i0 * 8 + j0] * w00 + gx[i0 * 8 + j1] * w01 +
              gx[i1 * 8 + j0] * w10 + gx[i1 * 8 + j1] * w11;
  float syv = gy[i0 * 8 + j0] * w00 + gy[i0 * 8 + j1] * w01 +
              gy[i1 * 8 + j0] * w10 + gy[i1 * 8 + j1] * w11;
  float wx = syv + (float)j;
  float wy = sxv + (float)i;
  float x0f = floorf(wx), y0f = floorf(wy);
  float dx = wx - x0f, dy = wy - y0f;
  int x0 = (int)x0f, y0 = (int)y0f;
  const float* im = data + (size_t)b * H * W;
  float v00 = sampz(im, x0, y0), v10 = sampz(im, x0 + 1, y0);
  float v01 = sampz(im, x0, y0 + 1), v11 = sampz(im, x0 + 1, y0 + 1);
  out[idx] = v00 * (1.0f - dx) * (1.0f - dy) + v10 * dx * (1.0f - dy) +
             v01 * (1.0f - dx) * dy + v11 * dx * dy;
}

// ---------------------------------------------------------------------------
// Host orchestration
// ---------------------------------------------------------------------------
static inline void cgemm(hipStream_t s, bool f16out, bool forward,
                         const _Float16* Ar, const _Float16* Ai, size_t asb, int lda,
                         const _Float16* Cb, const _Float16* Sb, const _Float16* Snb,
                         int ldb, void* Dr, void* Di, size_t dsb, int ldd,
                         int M, int N, int K, int batch, float scale) {
  // forward: effB = C - iS -> Bs1 = +S, Bs2 = -S ; inverse: Bs1 = -S, Bs2 = +S
  const _Float16* Bs1 = forward ? Sb : Snb;
  const _Float16* Bs2 = forward ? Snb : Sb;
  const bool cplxA = (Ai != nullptr);
  const bool cplxD = (Di != nullptr);
  dim3 g(N / 32, M / 32, batch), blk(32);
#define LAUNCH(F16, CA, CD)                                             \
  k_cgemm<F16, CA, CD><<<g, blk, 0, s>>>(Ar, Ai, asb, lda, Cb, Bs1, Bs2, \
                                         ldb, Dr, Di, dsb, ldd, K, scale)
  if (f16out) {
    if (cplxA) { if (cplxD) LAUNCH(true, true, true); else LAUNCH(true, true, false); }
    else       { if (cplxD) LAUNCH(true, false, true); else LAUNCH(true, false, false); }
  } else {
    if (cplxA) { if (cplxD) LAUNCH(false, true, true); else LAUNCH(false, true, false); }
    else       { if (cplxD) LAUNCH(false, false, true); else LAUNCH(false, false, false); }
  }
#undef LAUNCH
}

extern "C" void kernel_launch(void* const* d_in, const int* in_sizes, int n_in,
                              void* d_out, int out_size, void* d_ws, size_t ws_size,
                              hipStream_t stream) {
  (void)in_sizes; (void)n_in; (void)out_size; (void)ws_size;
  const float* data = (const float*)d_in[0];   // 32 x 768 x 768
  const float* tmpl = (const float*)d_in[1];   // 768 x 768
  float* out = (float*)d_out;
  float* ws = (float*)d_ws;

  const int B = 32, H = 768, W = 768, P = 128, T = 64;
  const size_t IMG = (size_t)H * W;
  const size_t BIMG = (size_t)B * IMG;
  const size_t PP = (size_t)P * P;

  float* o_pw = out;
  float* o_rigid = out + BIMG;
  float* o_rs = out + 2 * BIMG;
  float* o_sx = o_rs + 2 * (size_t)B;
  float* o_sy = o_sx + (size_t)B * 64;

  // Workspace arena (64B-aligned carve-outs)
  size_t off = 0;
#define RND(x) (((x) + 15) & ~(size_t)15)
#define ALLOC_F(nm, cnt) float* nm = ws + off; off += RND(cnt)
#define ALLOC_H(nm, cnt) _Float16* nm = (_Float16*)(ws + off); off += RND(((cnt) + 1) / 2)
  ALLOC_H(Cf768, IMG); ALLOC_H(Sf768, IMG); ALLOC_H(Nf768, IMG);
  ALLOC_H(Cf128, PP);  ALLOC_H(Sf128, PP);  ALLOC_H(Nf128, PP);
  ALLOC_H(data16, BIMG);
  ALLOC_H(tmpl16, IMG);
  ALLOC_H(tplPat16, (size_t)T * PP);
  ALLOC_F(FtR, IMG); ALLOC_F(FtI, IMG);
  ALLOC_F(tpfR, (size_t)T * PP); ALLOC_F(tpfI, (size_t)T * PP);
  ALLOC_F(rsW, 2 * B);
  ALLOC_F(sxW, (size_t)B * 64); ALLOC_F(syW, (size_t)B * 64);
  ALLOC_H(H0r, BIMG); ALLOC_H(H0i, BIMG);    // f16 intermediate pair 0
  ALLOC_H(H1r, BIMG); ALLOC_H(H1i, BIMG);    // f16 intermediate pair 1
  ALLOC_F(Fr, BIMG); ALLOC_F(Fi, BIMG);      // Fd (f32, kept for phase stage)
  ALLOC_F(Cr, BIMG); ALLOC_F(Ci, BIMG);      // f32 results (corr / patF / pcorr)
#undef ALLOC_F
#undef ALLOC_H
#undef RND

  const float invHW = 1.0f / (float)(H * W);
  const float invPP = 1.0f / (float)(P * P);

  // 1) DFT bases + f16 copies of inputs
  k_build_dft16<<<(int)((IMG + 255) / 256), 256, 0, stream>>>(Cf768, Sf768, Nf768, H, (int)IMG);
  k_build_dft16<<<(int)((PP + 255) / 256), 256, 0, stream>>>(Cf128, Sf128, Nf128, P, (int)PP);
  k_f32_to_f16<<<(int)((BIMG + 255) / 256), 256, 0, stream>>>(data, data16, BIMG);
  k_f32_to_f16<<<(int)((IMG + 255) / 256), 256, 0, stream>>>(tmpl, tmpl16, IMG);

  // 2) Template FFT: pass1 (real -> H0 f16), pass2 (H0 -> Ft f32)
  cgemm(stream, true, true, tmpl16, nullptr, 0, W, Cf768, Sf768, Nf768, W,
        H0r, H0i, 0, H, H, W, W, 1, 1.0f);
  cgemm(stream, false, true, H0r, H0i, 0, H, Cf768, Sf768, Nf768, H,
        FtR, FtI, 0, W, W, H, H, 1, 1.0f);

  // 3) Template patches + patch FFTs -> tpf (f32)
  k_extract_patches16<<<(int)(((size_t)T * PP + 255) / 256), 256, 0, stream>>>(
      tmpl, tplPat16, 0, (size_t)T * PP);
  cgemm(stream, true, true, tplPat16, nullptr, PP, P, Cf128, Sf128, Nf128, P,
        H0r, H0i, PP, P, P, P, P, T, 1.0f);
  cgemm(stream, false, true, H0r, H0i, PP, P, Cf128, Sf128, Nf128, P,
        tpfR, tpfI, PP, P, P, P, P, T, 1.0f);

  // 4) Data FFT -> Fd (f32 pair F)
  cgemm(stream, true, true, data16, nullptr, IMG, W, Cf768, Sf768, Nf768, W,
        H0r, H0i, IMG, H, H, W, W, B, 1.0f);
  cgemm(stream, false, true, H0r, H0i, IMG, H, Cf768, Sf768, Nf768, H,
        Fr, Fi, IMG, W, W, H, H, B, 1.0f);

  // 5) Cross power (normalized early, f16 out) -> H0
  k_crosspow16<<<(int)((BIMG + 255) / 256), 256, 0, stream>>>(
      Fr, Fi, FtR, FtI, H0r, H0i, IMG, 1, invHW, BIMG);

  // 6) corr = ifft2 -> C (f32 pair)
  cgemm(stream, true, false, H0r, H0i, IMG, W, Cf768, Sf768, Nf768, W,
        H1r, H1i, IMG, H, H, W, W, B, 1.0f);
  cgemm(stream, false, false, H1r, H1i, IMG, H, Cf768, Sf768, Nf768, H,
        Cr, Ci, IMG, W, W, H, H, B, 1.0f);

  // 7) Rigid subpixel shifts
  k_rigid_subpixel<<<B, 128, 0, stream>>>(Cr, Ci, rsW, o_rs);

  // 8) G = Fd * phase (f16) -> H0 ; data_rigid = Re(ifft2(G)) -> out
  k_phase_mult16<<<(int)((BIMG + 255) / 256), 256, 0, stream>>>(
      Fr, Fi, rsW, H0r, H0i, H, W, invHW, BIMG);
  cgemm(stream, true, false, H0r, H0i, IMG, W, Cf768, Sf768, Nf768, W,
        H1r, H1i, IMG, H, H, W, W, B, 1.0f);
  cgemm(stream, false, false, H1r, H1i, IMG, H, Cf768, Sf768, Nf768, H,
        o_rigid, nullptr, IMG, W, W, H, H, B, 1.0f);

  // 9) Patch correlation, chunked (8 frames = 512 patches per chunk).
  //    Reuse: pat16->H0r, fwd pair->H1, crosspow out->H0, inv1 out->hp1 (F region),
  //    patF / pcorr f32 pair -> C.
  _Float16* hp1r = (_Float16*)Fr;
  _Float16* hp1i = (_Float16*)Fi;
  const int BC = 8;
  for (int b0 = 0; b0 < B; b0 += BC) {
    const int nb = BC * T;
    const size_t tot = (size_t)nb * PP;
    k_extract_patches16<<<(int)((tot + 255) / 256), 256, 0, stream>>>(
        data, (_Float16*)H0r, b0, tot);
    cgemm(stream, true, true, (_Float16*)H0r, nullptr, PP, P, Cf128, Sf128, Nf128, P,
          H1r, H1i, PP, P, P, P, P, nb, 1.0f);
    cgemm(stream, false, true, H1r, H1i, PP, P, Cf128, Sf128, Nf128, P,
          Cr, Ci, PP, P, P, P, P, nb, 1.0f);
    k_crosspow16<<<(int)((tot + 255) / 256), 256, 0, stream>>>(
        Cr, Ci, tpfR, tpfI, H0r, H0i, PP, 64, invPP, tot);
    cgemm(stream, true, false, H0r, H0i, PP, P, Cf128, Sf128, Nf128, P,
          hp1r, hp1i, PP, P, P, P, P, nb, 1.0f);
    cgemm(stream, false, false, hp1r, hp1i, PP, P, Cf128, Sf128, Nf128, P,
          Cr, Ci, PP, P, P, P, P, nb, 1.0f);
    k_patch_subpixel<<<nb, 128, 0, stream>>>(Cr, Ci, rsW, sxW, syW, o_sx, o_sy, b0);
  }

  // 10) Upsample shift fields + piecewise warp -> data_pw
  k_resize_warp<<<(int)((BIMG + 255) / 256), 256, 0, stream>>>(
      data, sxW, syW, o_pw, BIMG);
}
